// GraphConv_38147899523081
// MI455X (gfx1250) — matrix operand order repaired and checked by reading, not compiled, not run on hardware
//
#include <hip/hip_runtime.h>

typedef _Float16 v16h __attribute__((ext_vector_type(16)));
typedef _Float16 v8h  __attribute__((ext_vector_type(8)));
typedef _Float16 v4h  __attribute__((ext_vector_type(4)));
typedef float    v8f  __attribute__((ext_vector_type(8)));

#define MROWS 6400   // B*N
#define NCOL  1024   // MID
#define KDIM  2048   // F
#define BM 128
#define BN 128
#define BK 64
#define LDST 72      // LDS row stride in halves (144B rows -> 16B aligned)

// Per-lane async copy of 16 bytes global -> LDS (ASYNCcnt-tracked, no VGPR staging).
// INST_OFFSET is added to BOTH the global and the LDS address (ISA 08 §4.4).
#define ASYNC_LD_B128(ldsaddr, gaddr, OFF)                                 \
  asm volatile("global_load_async_to_lds_b128 %0, %1, off offset:" #OFF   \
               :: "v"(ldsaddr), "v"(gaddr) : "memory")

#define WAIT_ASYNCCNT_0() asm volatile("s_wait_asynccnt 0" ::: "memory")

// ---------------- f32 -> f16 conversion ----------------
__global__ __launch_bounds__(256) void f32_to_f16_kernel(const float* __restrict__ in,
                                                         _Float16* __restrict__ out, int n) {
  int i = 4 * (blockIdx.x * blockDim.x + threadIdx.x);
  if (i < n) {
    float4 f = *(const float4*)(in + i);
    v4h h;
    h[0] = (_Float16)f.x; h[1] = (_Float16)f.y;
    h[2] = (_Float16)f.z; h[3] = (_Float16)f.w;
    *(v4h*)(out + i) = h;
  }
}

// A fragment (16x32 f16): lane<16 -> K {0..7,16..23}; lane>=16 -> K {8..15,24..31}
__device__ __forceinline__ v16h load_fragA(const _Float16* rowPtr, int kk, int hi) {
  const _Float16* p = rowPtr + kk + hi * 8;
  v8h lo = *(const v8h*)(p);
  v8h hh = *(const v8h*)(p + 16);
  v16h r;
#pragma unroll
  for (int i = 0; i < 8; ++i) { r[i] = lo[i]; r[i + 8] = hh[i]; }
  return r;
}

// B fragment (32x16 f16): lane = output column; lane<16 -> K {0..7,16..23}; lane>=16 -> K {8..15,24..31}
__device__ __forceinline__ v16h load_fragB(const _Float16* rowPtr, int kk, int hi) {
  const _Float16* p = rowPtr + kk + hi * 8;
  v8h lo = *(const v8h*)(p);
  v8h hh = *(const v8h*)(p + 16);
  v16h r;
#pragma unroll
  for (int i = 0; i < 8; ++i) { r[i] = lo[i]; r[i + 8] = hh[i]; }
  return r;
}

// ---------------- P = V @ W^T via WMMA f16, async double-buffered LDS ----------------
__global__ __launch_bounds__(256) void gemm_f16_wmma(const _Float16* __restrict__ A,
                                                     const _Float16* __restrict__ Bm,
                                                     float* __restrict__ P) {
  __shared__ _Float16 sA[2][BM * LDST];
  __shared__ _Float16 sB[2][BN * LDST];

  const int tid = threadIdx.x;
  const int mBase = blockIdx.y * BM;
  const int nBase = blockIdx.x * BN;
  const int wave = tid >> 5;
  const int lane = tid & 31;
  const int laneLow = lane & 15;
  const int hi = lane >> 4;
  const int wm = wave & 3;   // 4 waves along M (32 rows each)
  const int wn = wave >> 2;  // 2 waves along N (64 cols each)

  const int ldRow = tid >> 1;        // 0..127
  const int ldCol = (tid & 1) * 32;  // 0 or 32 (halves)
  const _Float16* gA = A + (size_t)(mBase + ldRow) * KDIM + ldCol;
  const _Float16* gB = Bm + (size_t)(nBase + ldRow) * KDIM + ldCol;

  // LDS byte addresses for this thread's staging slice (generic ptr low 32 bits = LDS offset)
  const unsigned ldsA0 = (unsigned)(size_t)&sA[0][ldRow * LDST + ldCol];
  const unsigned ldsA1 = (unsigned)(size_t)&sA[1][ldRow * LDST + ldCol];
  const unsigned ldsB0 = (unsigned)(size_t)&sB[0][ldRow * LDST + ldCol];
  const unsigned ldsB1 = (unsigned)(size_t)&sB[1][ldRow * LDST + ldCol];

  v8f acc[2][4] = {};

  // issue one BK-slice (32 halves = 4x b128 per matrix per thread) into buffer `buf`
  auto issue = [&](int buf, int k0) {
    const unsigned long long a = (unsigned long long)(size_t)(gA + k0);
    const unsigned long long b = (unsigned long long)(size_t)(gB + k0);
    const unsigned la = buf ? ldsA1 : ldsA0;
    const unsigned lb = buf ? ldsB1 : ldsB0;
    ASYNC_LD_B128(la, a, 0);
    ASYNC_LD_B128(la, a, 16);
    ASYNC_LD_B128(la, a, 32);
    ASYNC_LD_B128(la, a, 48);
    ASYNC_LD_B128(lb, b, 0);
    ASYNC_LD_B128(lb, b, 16);
    ASYNC_LD_B128(lb, b, 32);
    ASYNC_LD_B128(lb, b, 48);
  };

  const int NIT = KDIM / BK;  // 32
  issue(0, 0);

  for (int it = 0; it < NIT; ++it) {
    WAIT_ASYNCCNT_0();   // this wave's loads for buffer (it&1) landed in LDS
    __syncthreads();     // all waves' loads landed; all waves done reading buffer !(it&1)
    if (it + 1 < NIT) issue((it + 1) & 1, (it + 1) * BK);  // overlap next slice with compute

    const _Float16* cA = &sA[it & 1][0];
    const _Float16* cB = &sB[it & 1][0];
#pragma unroll
    for (int kk = 0; kk < BK; kk += 32) {
      v16h af[2];
#pragma unroll
      for (int ti = 0; ti < 2; ++ti)
        af[ti] = load_fragA(cA + (wm * 32 + ti * 16 + laneLow) * LDST, kk, hi);
      v16h bf[4];
#pragma unroll
      for (int tj = 0; tj < 4; ++tj)
        bf[tj] = load_fragB(cB + (wn * 64 + tj * 16 + laneLow) * LDST, kk, hi);
#pragma unroll
      for (int ti = 0; ti < 2; ++ti)
#pragma unroll
        for (int tj = 0; tj < 4; ++tj)
          acc[ti][tj] = __builtin_amdgcn_wmma_f32_16x16x32_f16(
              false, af[ti], false, bf[tj], (short)0, acc[ti][tj], false, false);
    }
    // frag ds_loads are consumed before this wave reaches the next barrier, so the
    // buffer can be refilled after that barrier: one barrier per iteration suffices.
  }

#pragma unroll
  for (int ti = 0; ti < 2; ++ti)
#pragma unroll
    for (int tj = 0; tj < 4; ++tj) {
      const int r0 = mBase + wm * 32 + ti * 16 + hi * 8;
      const int c0 = nBase + wn * 64 + tj * 16 + laneLow;
      float* q = P + (size_t)r0 * NCOL + c0;
#pragma unroll
      for (int r = 0; r < 8; ++r) q[(size_t)r * NCOL] = acc[ti][tj][r];
    }
}

// ---------------- gather + gaussian-weighted aggregation ----------------
__global__ __launch_bounds__(256) void aggregate_kernel(
    const float* __restrict__ P, const float* __restrict__ coord,
    const float* __restrict__ adj, const int* __restrict__ top,
    const float* __restrict__ mean_rho, const float* __restrict__ mean_theta,
    const float* __restrict__ prec_rho, const float* __restrict__ prec_theta,
    float* __restrict__ out) {
  __shared__ float s_w[16][8];
  __shared__ int s_row[16];

  const int bn = blockIdx.x;  // b*100 + n
  const int b = bn / 100;
  const int tid = threadIdx.x;

  if (tid < 16) {
    const int k = tid;
    const int j = top[bn * 16 + k];
    const float a = adj[bn * 16 + k];
    const float rho   = coord[((size_t)bn * 100 + j) * 2 + 0];
    const float theta = coord[((size_t)bn * 100 + j) * 2 + 1];
    float wv[8];
    float wsum = 0.f;
#pragma unroll
    for (int m = 0; m < 8; ++m) {
      const float mr = mean_rho[m], pr = prec_rho[m];
      const float mt = mean_theta[m], pt = prec_theta[m];
      const float dr = rho - mr;
      const float wr = __expf(-0.5f * dr * dr / (1e-14f + pr * pr));
      const float fa = fabsf(theta - mt);
      const float sa = fabsf(6.2831853071795864769f - fa);
      const float d = fminf(fa, sa);
      const float wt = __expf(-0.5f * d * d / (1e-14f + pt * pt));
      float w = wr * wt;
      if (!(w == w)) w = 0.f;  // NaN -> 0
      wv[m] = w;
      wsum += w;
    }
    const float s = a / (wsum + 1e-14f);  // fold adjacency into normalized weight
#pragma unroll
    for (int m = 0; m < 8; ++m) s_w[k][m] = wv[m] * s;
    s_row[k] = b * 100 + j;
  }
  __syncthreads();

  const int c = tid * 4;      // 4 contiguous output columns per thread
  const int m = tid >> 5;     // kernel index = wave id (uniform per wave)
  float a0 = 0.f, a1 = 0.f, a2 = 0.f, a3 = 0.f;
#pragma unroll
  for (int k = 0; k < 16; ++k) {
    const float w = s_w[k][m];
    const float4 p = *(const float4*)(P + (size_t)s_row[k] * NCOL + c);
    a0 += w * p.x; a1 += w * p.y; a2 += w * p.z; a3 += w * p.w;
  }
  float4 o; o.x = a0; o.y = a1; o.z = a2; o.w = a3;
  *(float4*)(out + (size_t)bn * NCOL + c) = o;
}

extern "C" void kernel_launch(void* const* d_in, const int* in_sizes, int n_in,
                              void* d_out, int out_size, void* d_ws, size_t ws_size,
                              hipStream_t stream) {
  const float* v     = (const float*)d_in[0];
  // d_in[1] = v_mask (unused by the reference computation)
  const float* coord = (const float*)d_in[2];
  const float* adj   = (const float*)d_in[3];
  const int*   top   = (const int*)d_in[4];
  const float* mrho  = (const float*)d_in[5];
  const float* mth   = (const float*)d_in[6];
  const float* prho  = (const float*)d_in[7];
  const float* pth   = (const float*)d_in[8];
  const float* W     = (const float*)d_in[9];

  // workspace layout: Vh (f16) | Wh (f16) | P (f32)
  _Float16* Vh = (_Float16*)d_ws;
  _Float16* Wh = Vh + (size_t)MROWS * KDIM;
  float* P = (float*)((char*)d_ws + (size_t)MROWS * KDIM * 2 + (size_t)NCOL * KDIM * 2);

  f32_to_f16_kernel<<<(MROWS * KDIM / 4 + 255) / 256, 256, 0, stream>>>(v, Vh, MROWS * KDIM);
  f32_to_f16_kernel<<<(NCOL * KDIM / 4 + 255) / 256, 256, 0, stream>>>(W, Wh, NCOL * KDIM);

  dim3 grid(NCOL / BN, MROWS / BM);  // (8, 50)
  gemm_f16_wmma<<<grid, 256, 0, stream>>>(Vh, Wh, P);

  aggregate_kernel<<<MROWS, 256, 0, stream>>>(P, coord, adj, top, mrho, mth, prho, pth,
                                              (float*)d_out);
}
